// EncoderRNN_58377195487981
// MI455X (gfx1250) — compile-verified, matrix operand order checked by reading, not run on hardware
//
#include <hip/hip_runtime.h>

// EncoderRNN (GRU, seq=512, H=1024) for MI455X / gfx1250 (wave32).
//
// Phase 1: gi = emb[tokens] @ w_ih^T + b_ih  via V_WMMA_F32_16X16X4_F32
//          (exact fp32 WMMA; GEMM is tiny vs. the scan).
// Phase 2: persistent-kernel sequential GRU scan. 64 WGs; each WG caches its
//          48 rows of w_hh (192 KB) in LDS (320 KB/WGP budget) — streamed in
//          via the Tensor Data Mover when available — stages h in LDS each
//          step via async-to-LDS, grid-syncs via an agent-scope atomic counter.

#define SEQ   512
#define HID   1024
#define G3    3072
#define NWG   64

typedef __attribute__((ext_vector_type(8))) float v8f;
typedef __attribute__((ext_vector_type(2))) float v2f;
typedef unsigned int u32;
typedef __attribute__((ext_vector_type(4))) u32 u32x4;
typedef __attribute__((ext_vector_type(8))) int i32x8;
typedef __attribute__((ext_vector_type(4))) int i32x4;

// Address-space-qualified vector pointees for the async-to-LDS builtin
// (param types per hipcc diagnostic: v4i in AS1 / AS3, non-const).
typedef __attribute__((address_space(1))) i32x4 gvec_t;
typedef __attribute__((address_space(3))) i32x4 lvec_t;

#if defined(__has_builtin)
#  if __has_builtin(__builtin_amdgcn_wmma_f32_16x16x4_f32)
#    define HAVE_WMMA_F32 1
#  endif
   // TDM builtin arity differs by toolchain: ROCm 7.2 (clang-22) = 5 args;
   // amdgpu-toolchain (clang-23 + therock headers, ships amd_gfx1250_TDM.h)
   // = 6 args. Use the header as the discriminator.
#  if __has_builtin(__builtin_amdgcn_tensor_load_to_lds) && \
      __has_builtin(__builtin_amdgcn_s_wait_tensorcnt)
#    if __has_include(<hip/amd_detail/amd_gfx1250_TDM.h>)
#      define HAVE_TDM6 1
#    else
#      define HAVE_TDM5 1
#    endif
#  endif
#  if __has_builtin(__builtin_amdgcn_global_load_async_to_lds_b128) && \
      __has_builtin(__builtin_amdgcn_s_wait_asynccnt)
#    define HAVE_ASYNC_LDS 1
#  endif
#endif

#if !defined(HAVE_WMMA_F32)
typedef __attribute__((ext_vector_type(16))) __bf16 v16bf;
#endif

// ---------------------------------------------------------------- init ------
__global__ __launch_bounds__(256) void gru_init(float* __restrict__ hbuf,
                                                unsigned* __restrict__ ctr) {
    for (int i = threadIdx.x; i < 2 * HID; i += 256) hbuf[i] = 0.0f;
    if (threadIdx.x == 0) *ctr = 0u;
}

// ------------------------------------------------------- phase 1: WMMA GEMM -
// gi[m][n] = sum_k emb[tok[m]][k] * w_ih[n][k] + b_ih[n]
// One wave (32 lanes) per 16x16 output tile. 8 waves/block along N.
__global__ __launch_bounds__(256) void gi_gemm(const int*   __restrict__ tok,
                                               const float* __restrict__ emb,
                                               const float* __restrict__ w_ih,
                                               const float* __restrict__ b_ih,
                                               float*       __restrict__ gi) {
    const int lane   = threadIdx.x & 31;
    const int wave   = threadIdx.x >> 5;
    const int n0     = (blockIdx.x * 8 + wave) * 16;  // N tile base (0..3056)
    const int m0     = blockIdx.y * 16;               // M tile base (0..496)
    const int l15    = lane & 15;
    const int hihalf = lane >> 4;                     // 0: lanes 0-15, 1: 16-31

    const float* arow = emb  + (size_t)tok[m0 + l15] * HID;  // A: M = l15
    const float* brow = w_ih + (size_t)(n0 + l15)    * HID;  // B^T row: N = l15

    v8f acc;
    const float bias = b_ih[n0 + l15];
#pragma unroll
    for (int r = 0; r < 8; ++r) acc[r] = bias;

#if defined(HAVE_WMMA_F32)
    // f32 A 16x4 layout: lanes 0-15 hold K={k0,k0+1}, lanes 16-31 K={k0+2,k0+3}
    const int koff = hihalf * 2;
#pragma unroll 8
    for (int k0 = 0; k0 < HID; k0 += 4) {
        v2f a = *(const v2f*)(arow + k0 + koff);
        v2f b = *(const v2f*)(brow + k0 + koff);
        acc = __builtin_amdgcn_wmma_f32_16x16x4_f32(
            false, a, false, b, (short)0, acc, false, false);
    }
#else
    // Fallback: bf16 WMMA 16x16x32 with on-the-fly f32->bf16 conversion.
    const int kb = hihalf << 3;
#pragma unroll 4
    for (int k0 = 0; k0 < HID; k0 += 32) {
        v16bf a, b;
#pragma unroll
        for (int idx = 0; idx < 16; ++idx) {
            int k = k0 + ((idx >> 3) << 4) + kb + (idx & 7);
            a[idx] = (__bf16)arow[k];
            b[idx] = (__bf16)brow[k];
        }
        acc = __builtin_amdgcn_wmma_f32_16x16x32_bf16(
            false, a, false, b, (short)0, acc, false, false);
    }
#endif

    // C/D f32 16x16: VGPR r -> M = m0 + r + 8*hihalf, N = n0 + l15
    const int col = n0 + l15;
    const int mb  = m0 + (hihalf << 3);
#pragma unroll
    for (int r = 0; r < 8; ++r)
        gi[(size_t)(mb + r) * G3 + col] = acc[r];
}

// -------------------------------------------------- phase 2: persistent scan -
// 64 WGs x 256 threads. WG g owns h[g*16 .. g*16+16) and the 48 w_hh rows
// {gate*1024 + g*16 + i : gate in 0..2, i in 0..15}, resident in LDS.
__global__ __launch_bounds__(256) void gru_scan(const float* __restrict__ w_hh,
                                                const float* __restrict__ b_hh,
                                                const float* __restrict__ gi,
                                                float*       __restrict__ hbuf,
                                                unsigned*    __restrict__ ctr,
                                                float*       __restrict__ out) {
    extern __shared__ float smem[];
    float* lds_w = smem;                 // 48 * 1024 floats (192 KB)
    float* lds_h = smem + 48 * HID;      // 1024 floats
    float* lds_d = lds_h + HID;          // 48 dot results (+pad)

    const int tid  = threadIdx.x;
    const int lane = tid & 31;
    const int wave = tid >> 5;
    const int wg   = blockIdx.x;

    // ---- preload this WG's 48 rows of w_hh into LDS
#if defined(HAVE_TDM5) || defined(HAVE_TDM6)
    // Tensor Data Mover: three 2D tiles (16 rows x 1024 f32, row pitch 1024)
    // DMA'd straight into LDS. Dynamic LDS starts at offset 0 here (no static
    // LDS in this kernel), so lds_addr = gate * 64KB.
    if (wave == 0) {
#pragma unroll
        for (int gate = 0; gate < 3; ++gate) {
            const unsigned long long ga = (unsigned long long)
                (const void*)(w_hh + ((size_t)(gate * HID + wg * 16)) * HID);
            u32x4 g0;
            g0[0] = 1u;                                   // count=1, user mode
            g0[1] = (u32)(gate * 16 * HID * 4);           // lds_addr (bytes)
            g0[2] = (u32)ga;                              // global_addr[31:0]
            g0[3] = (u32)((ga >> 32) & 0x01ffffffu) | (2u << 30); // [56:32]|type=2
            const u32 td0 = HID, td1 = G3, tile0 = HID, tile1 = 16, s0 = HID;
            i32x8 g1;
            g1[0] = (int)(2u << 16);                      // data_size=4B, mask=0
            g1[1] = (int)((td0 & 0xffffu) << 16);         // tensor_dim0 lo16
            g1[2] = (int)((td0 >> 16) | ((td1 & 0xffffu) << 16));
            g1[3] = (int)((td1 >> 16) | (tile0 << 16));   // tile_dim0
            g1[4] = (int)tile1;                           // tile_dim1; dim2=0
            g1[5] = (int)s0;                              // dim0_stride lo32
            g1[6] = 0;
            g1[7] = 0;
            i32x4 gz4 = {0, 0, 0, 0};
#if defined(HAVE_TDM5)
            __builtin_amdgcn_tensor_load_to_lds(g0, g1, gz4, gz4, 0);
#else
            i32x8 gz8 = {0, 0, 0, 0, 0, 0, 0, 0};
            __builtin_amdgcn_tensor_load_to_lds(g0, g1, gz4, gz4, gz8, 0);
#endif
        }
        __builtin_amdgcn_s_wait_tensorcnt(0);
    }
#else
    for (int idx = tid; idx < 48 * 256; idx += 256) {
        const int j = idx >> 8;          // LDS row 0..47 (= gate*16 + i)
        const int c = idx & 255;         // float4 column
        const int grow = (j >> 4) * HID + wg * 16 + (j & 15);
        *(float4*)(lds_w + j * HID + c * 4) =
            *(const float4*)(w_hh + (size_t)grow * HID + c * 4);
    }
#endif
    float bhr = 0.f, bhz = 0.f, bhn = 0.f;
    const int gidx = wg * 16 + tid;      // valid for tid < 16
    if (tid < 16) {
        bhr = b_hh[gidx];
        bhz = b_hh[HID + gidx];
        bhn = b_hh[2 * HID + gidx];
    }
    __syncthreads();

    for (int t = 0; t < SEQ; ++t) {
        // ---- stage current h (written last step by all WGs; visible via barrier)
        const float* hcur = hbuf + (t & 1) * HID;
#if defined(HAVE_ASYNC_LDS)
        __builtin_amdgcn_global_load_async_to_lds_b128(
            (gvec_t*)(hcur + tid * 4),
            (lvec_t*)(lds_h + tid * 4),
            0, 0);
        __builtin_amdgcn_s_wait_asynccnt(0);
#else
        *(float4*)(lds_h + tid * 4) = *(const float4*)(hcur + tid * 4);
#endif
        __syncthreads();

        // ---- 48 dot products of length 1024; wave w handles rows 6w..6w+5
        const float4* hv = (const float4*)lds_h;
#pragma unroll
        for (int jr = 0; jr < 6; ++jr) {
            const int j = wave * 6 + jr;
            const float4* wv = (const float4*)(lds_w + j * HID);
            float s = 0.f;
#pragma unroll
            for (int it = 0; it < 8; ++it) {
                const float4 a = wv[lane + 32 * it];
                const float4 b = hv[lane + 32 * it];
                s += a.x * b.x + a.y * b.y + a.z * b.z + a.w * b.w;
            }
#pragma unroll
            for (int off = 16; off > 0; off >>= 1)
                s += __shfl_xor(s, off, 32);
            if (lane == 0) lds_d[j] = s;
        }
        __syncthreads();

        // ---- gate math for this WG's 16 h elements
        if (tid < 16) {
            const float ghr = lds_d[tid]      + bhr;
            const float ghz = lds_d[16 + tid] + bhz;
            const float ghn = lds_d[32 + tid] + bhn;
            const float* grow = gi + (size_t)t * G3;
            const float r = 1.f / (1.f + __expf(-(grow[gidx] + ghr)));
            const float z = 1.f / (1.f + __expf(-(grow[HID + gidx] + ghz)));
            const float n = tanhf(grow[2 * HID + gidx] + r * ghn);
            const float hn = (1.f - z) * n + z * lds_h[gidx];
            hbuf[((t + 1) & 1) * HID + gidx] = hn;
            out[(size_t)t * HID + gidx] = hn;
            if (t == SEQ - 1) out[(size_t)SEQ * HID + gidx] = hn;  // h_last
        }

        // ---- grid barrier: release h writes, arrive, spin, acquire
        __threadfence();
        __syncthreads();
        if (tid == 0) {
            __hip_atomic_fetch_add(ctr, 1u, __ATOMIC_RELEASE,
                                   __HIP_MEMORY_SCOPE_AGENT);
            const unsigned target = (unsigned)NWG * (unsigned)(t + 1);
            while (__hip_atomic_load(ctr, __ATOMIC_ACQUIRE,
                                     __HIP_MEMORY_SCOPE_AGENT) < target)
                __builtin_amdgcn_s_sleep(2);
        }
        __syncthreads();
    }
}

// --------------------------------------------------------------- launcher ---
extern "C" void kernel_launch(void* const* d_in, const int* in_sizes, int n_in,
                              void* d_out, int out_size, void* d_ws, size_t ws_size,
                              hipStream_t stream) {
    const int*   tok  = (const int*)  d_in[0];   // (1, 512)
    const float* emb  = (const float*)d_in[1];   // (30000, 1024)
    const float* w_ih = (const float*)d_in[2];   // (3072, 1024)
    const float* w_hh = (const float*)d_in[3];   // (3072, 1024)
    const float* b_ih = (const float*)d_in[4];   // (3072,)
    const float* b_hh = (const float*)d_in[5];   // (3072,)
    float* out = (float*)d_out;                  // 512*1024 outputs + 1024 h_last

    float*    gi   = (float*)d_ws;                       // 512*3072 f32
    float*    hbuf = gi + (size_t)SEQ * G3;              // 2*1024 f32 (double buffer)
    unsigned* ctr  = (unsigned*)(hbuf + 2 * HID);        // grid-barrier counter

    gru_init<<<dim3(1), dim3(256), 0, stream>>>(hbuf, ctr);

    // 192 N-tiles (24 blocks x 8 waves) x 32 M-tiles
    gi_gemm<<<dim3(24, 32), dim3(256), 0, stream>>>(tok, emb, w_ih, b_ih, gi);

    const size_t smem_bytes = (size_t)(48 * HID + HID + 64) * sizeof(float); // ~196 KB
    gru_scan<<<dim3(NWG), dim3(256), smem_bytes, stream>>>(w_hh, b_hh, gi,
                                                           hbuf, ctr, out);
}
